// MultiScaleSpectralAttention_55448027791805
// MI455X (gfx1250) — matrix operand order repaired and checked by reading, not compiled, not run on hardware
//
#include <hip/hip_runtime.h>
#include <math.h>

#define BB 4
#define NN 4096
#define DMM 1024
#define HH 8
#define HDD 128
#define FGG 2049
#define AGC (HH*FGG*2)   /* 32784 */

typedef __attribute__((ext_vector_type(16))) _Float16 v16h;
typedef __attribute__((ext_vector_type(8)))  float    v8f;

__device__ __forceinline__ float gelu_f(float x) {
    return 0.5f * x * (1.0f + erff(x * 0.70710678118654752f));
}

// ---------------------------------------------------------------------------
// Kernel 1a: LayerNorm statistics per row (b,n): mean and rstd over DM=1024.
// ---------------------------------------------------------------------------
__global__ void __launch_bounds__(256) ln_stats(const float* __restrict__ x,
                                                float* __restrict__ mu,
                                                float* __restrict__ rstd) {
    int row = blockIdx.x;                       // b*N + n
    const float* xr = x + (size_t)row * DMM;
    float s = 0.0f, s2 = 0.0f;
    for (int i = threadIdx.x; i < DMM; i += 256) {
        float v = xr[i];
        s += v; s2 += v * v;
    }
    __shared__ float sh[256], sh2[256];
    sh[threadIdx.x] = s; sh2[threadIdx.x] = s2;
    __syncthreads();
    for (int o = 128; o > 0; o >>= 1) {
        if (threadIdx.x < o) {
            sh[threadIdx.x]  += sh[threadIdx.x + o];
            sh2[threadIdx.x] += sh2[threadIdx.x + o];
        }
        __syncthreads();
    }
    if (threadIdx.x == 0) {
        float m = sh[0] * (1.0f / DMM);
        float v = sh2[0] * (1.0f / DMM) - m * m;
        mu[row]   = m;
        rstd[row] = rsqrtf(v + 1e-5f);
    }
}

// ---------------------------------------------------------------------------
// Kernel 1b: apply LN affine and transpose into xnT (B, DM, N).
// Both global read and write coalesced via 32x33 LDS tile.
// ---------------------------------------------------------------------------
__global__ void __launch_bounds__(256) ln_transpose(const float* __restrict__ x,
                                                    const float* __restrict__ mu,
                                                    const float* __restrict__ rstd,
                                                    const float* __restrict__ g,
                                                    const float* __restrict__ beta,
                                                    float* __restrict__ xnT) {
    __shared__ float tile[32][33];
    int b  = blockIdx.z;
    int n0 = blockIdx.x * 32;
    int c0 = blockIdx.y * 32;
    int tx = threadIdx.x, ty = threadIdx.y;
    #pragma unroll
    for (int i = 0; i < 4; i++) {
        int n = n0 + ty * 4 + i;
        int c = c0 + tx;
        float v = x[((size_t)b * NN + n) * DMM + c];
        v = (v - mu[b * NN + n]) * rstd[b * NN + n] * g[c] + beta[c];
        tile[ty * 4 + i][tx] = v;
    }
    __syncthreads();
    #pragma unroll
    for (int i = 0; i < 4; i++) {
        int c = c0 + ty * 4 + i;
        int n = n0 + tx;
        xnT[((size_t)b * DMM + c) * NN + n] = tile[tx][ty * 4 + i];
    }
}

// ---------------------------------------------------------------------------
// Kernel 1c: ctx[b][c] = sum_n xnT[b][c][n]  (coalesced row sum, no atomics)
// ---------------------------------------------------------------------------
__global__ void __launch_bounds__(256) ctx_sum(const float* __restrict__ xnT,
                                               float* __restrict__ ctx) {
    int bc = blockIdx.x;                        // b*DM + c
    const float* row = xnT + (size_t)bc * NN;
    float s = 0.0f;
    for (int i = threadIdx.x; i < NN; i += 256) s += row[i];
    __shared__ float sh[256];
    sh[threadIdx.x] = s;
    __syncthreads();
    for (int o = 128; o > 0; o >>= 1) {
        if (threadIdx.x < o) sh[threadIdx.x] += sh[threadIdx.x + o];
        __syncthreads();
    }
    if (threadIdx.x == 0) ctx[bc] = sh[0];      // stored as SUM; scaled by 1/N in GEMM
}

// ---------------------------------------------------------------------------
// Kernel 2: generic 16-row GEMM on the WMMA path.
// Out(16 x Nw) = act( A(16 x K)*a_scale  @  W(K x Nw) + bias )
// One wave (32 threads) per 16-column tile; f16 operands, f32 accumulate.
// act: 0=none, 1=gelu, 2=sigmoid.
// Fragment layouts per CDNA5 ISA 7.12.2 (wave32).
// Edge guard is branchless (clamped index + 0/1 mask) so all fragment loads
// issue unconditionally — no per-element exec-mask regions.
// ---------------------------------------------------------------------------
__global__ void __launch_bounds__(32) gemm16_wmma(const float* __restrict__ A,
                                                  float a_scale,
                                                  const float* __restrict__ W,
                                                  const float* __restrict__ bias,
                                                  float* __restrict__ Out,
                                                  int K, int Nw, int act) {
    int lane = threadIdx.x;
    int m = lane & 15;          // A row / B-D column-within-tile
    int g = lane >> 4;          // lane group
    int col  = blockIdx.x * 16 + m;
    int colc = (col < Nw) ? col : (Nw - 1);     // clamped (branchless guard)
    float cmask = (col < Nw) ? 1.0f : 0.0f;

    v8f acc = {};
    for (int kb = 0; kb < K; kb += 32) {
        v16h a, bv;
        #pragma unroll
        for (int e = 0; e < 16; e++) {
            // A 16x32 f16 fragment: lane group selects K sub-block of 8
            int ka = kb + 16 * (e >> 3) + 8 * g + (e & 7);
            a[e] = (_Float16)(A[(size_t)m * K + ka] * a_scale);
            // B 32x16 f16 fragment: same K order as A (lane group selects K sub-block of 8)
            int kbi = kb + 16 * (e >> 3) + 8 * g + (e & 7);
            bv[e] = (_Float16)(W[(size_t)kbi * Nw + colc] * cmask);
        }
        if (kb + 32 < K)   // stream next weight panel (global_prefetch_b8)
            __builtin_prefetch(&W[(size_t)(kb + 32 + 16 * g) * Nw + colc], 0, 0);
        acc = __builtin_amdgcn_wmma_f32_16x16x32_f16(
            /*neg_a=*/false, a, /*neg_b=*/false, bv,
            /*c_mod=*/(short)0, acc, /*reuse_a=*/false, /*reuse_b=*/false);
    }
    if (col < Nw) {
        float bval = bias[col];
        #pragma unroll
        for (int v = 0; v < 8; v++) {
            int row = v + 8 * g;                 // D row per ISA C/D layout
            float val = acc[v] + bval;
            if (act == 1)      val = gelu_f(val);
            else if (act == 2) val = 1.0f / (1.0f + expf(-val));
            Out[(size_t)row * Nw + col] = val;
        }
    }
}

// ---------------------------------------------------------------------------
// Kernel 3: per (b, c=h*HD+hd): window + 4096-pt FFT in LDS, spectral filter,
// inverse FFT, Haar local branch, alpha-fused combine. Writes fusedT (B,DM,N).
// Column load uses GLOBAL_LOAD_ASYNC_TO_LDS_B128 (ASYNCcnt path, bypasses
// VGPRs). Forward DIF (natural->bitrev), filter in bitrev domain, inverse DIT
// (bitrev->natural): no permutation passes. ortho norm = 1/64 each direction.
// Spectrum stays Hermitian (eff/bias/mag-scale real) so full complex FFT
// real-part == irfft exactly. LDS: raw (16KB) + complex buf (32KB) = 48KB.
// ---------------------------------------------------------------------------
__global__ void __launch_bounds__(256) spectral_fuse(const float* __restrict__ xnT,
                                                     const float* __restrict__ bfg,
                                                     const float* __restrict__ bbg,
                                                     const float* __restrict__ bfl,
                                                     const float* __restrict__ bbl,
                                                     const float* __restrict__ ag,
                                                     const float* __restrict__ al,
                                                     const float* __restrict__ alphav,
                                                     float* __restrict__ fusedT) {
    extern __shared__ float smem[];
    float*  raw = smem;                          // NN floats
    float2* buf = (float2*)(smem + NN);          // NN float2

    int blk = blockIdx.x;                        // b*DM + c
    int b = blk / DMM;
    int c = blk % DMM;
    int h = c / HDD;
    const float* src = xnT + (size_t)blk * NN;
    int t = threadIdx.x;

    // ---- Async DMA of the column into LDS (16B per lane per issue) ----
    #pragma unroll
    for (int s = 0; s < 4; ++s) {
        int i4 = (t + s * 256) * 4;              // float index, 16B granule
        unsigned lds_addr = (unsigned)(uintptr_t)(raw + i4);  // low 32b = LDS offset
        const float* gp = src + i4;
        asm volatile("global_load_async_to_lds_b128 %0, %1, off"
                     :: "v"(lds_addr), "v"(gp) : "memory");
    }
    asm volatile("s_wait_asynccnt 0x0" ::: "memory");
    __syncthreads();

    // Windowed copy into FFT buffer (raw stays for the local branch).
    const float TWO_PI = 6.2831853071795864769f;
    for (int i = t; i < NN; i += 256) {
        float w = 0.5f - 0.5f * cosf(TWO_PI * (float)i / (float)(NN - 1));
        buf[i] = make_float2(raw[i] * w, 0.0f);
    }
    __syncthreads();

    // ---- Local (Haar pair) branch, in place on raw ----
    const float S2 = 0.70710678118654752f;
    float effl0 = bfl[h * 2 + 0] * (1.0f + al[b * 32 + (h * 2 + 0) * 2 + 0]);
    float ebl0  = bbl[h * 2 + 0] +         al[b * 32 + (h * 2 + 0) * 2 + 1];
    float effl1 = bfl[h * 2 + 1] * (1.0f + al[b * 32 + (h * 2 + 1) * 2 + 0]);
    float ebl1  = bbl[h * 2 + 1] +         al[b * 32 + (h * 2 + 1) * 2 + 1];
    for (int i = t; i < NN / 2; i += 256) {
        float p0 = raw[2 * i], p1 = raw[2 * i + 1];
        float a = (p0 + p1) * S2;
        float d = (p0 - p1) * S2;
        float am = gelu_f(a * effl0 + ebl0);
        float dm = gelu_f(d * effl1 + ebl1);
        raw[2 * i]     = (am + dm) * S2;
        raw[2 * i + 1] = (am - dm) * S2;
    }

    // ---- Forward FFT: radix-2 DIF, natural in -> bit-reversed out ----
    for (int sh = 11; sh >= 0; --sh) {
        __syncthreads();
        int half = 1 << sh;
        float invhalf = 1.0f / (float)half;
        for (int j = t; j < NN / 2; j += 256) {
            int pos = j & (half - 1);
            int i1  = ((j >> sh) << (sh + 1)) + pos;
            int i2  = i1 + half;
            float2 u = buf[i1], v = buf[i2];
            float ang = -3.14159265358979323846f * (float)pos * invhalf;
            float sn, cs; __sincosf(ang, &sn, &cs);
            buf[i1] = make_float2(u.x + v.x, u.y + v.y);
            float dr = u.x - v.x, di = u.y - v.y;
            buf[i2] = make_float2(dr * cs - di * sn, dr * sn + di * cs);
        }
    }
    __syncthreads();

    // ---- Spectral filter in bit-reversed domain ----
    const float ortho = 0.015625f;               // 1/sqrt(4096)
    for (int j = t; j < NN; j += 256) {
        int k   = (int)(__brev((unsigned)j) >> 20);
        int idx = (k <= NN / 2) ? k : (NN - k);  // mirror: Hermitian-consistent
        float eff  = bfg[h * FGG + idx] * (1.0f + ag[(size_t)b * AGC + (h * FGG + idx) * 2 + 0]);
        float bias = bbg[h * FGG + idx] +         ag[(size_t)b * AGC + (h * FGG + idx) * 2 + 1];
        float zr = buf[j].x * ortho * eff + bias;  // bias is real
        float zi = buf[j].y * ortho * eff;
        float mag = sqrtf(zr * zr + zi * zi);
        float gl  = gelu_f(mag) / (mag + 1e-6f);
        buf[j] = make_float2(zr * gl, zi * gl);
    }

    // ---- Inverse FFT: radix-2 DIT, bit-reversed in -> natural out ----
    for (int sh = 0; sh <= 11; ++sh) {
        __syncthreads();
        int half = 1 << sh;
        float invhalf = 1.0f / (float)half;
        for (int j = t; j < NN / 2; j += 256) {
            int pos = j & (half - 1);
            int i1  = ((j >> sh) << (sh + 1)) + pos;
            int i2  = i1 + half;
            float ang = 3.14159265358979323846f * (float)pos * invhalf;
            float sn, cs; __sincosf(ang, &sn, &cs);
            float2 u = buf[i1], v = buf[i2];
            float vr = v.x * cs - v.y * sn;
            float vi = v.x * sn + v.y * cs;
            buf[i1] = make_float2(u.x + vr, u.y + vi);
            buf[i2] = make_float2(u.x - vr, u.y - vi);
        }
    }
    __syncthreads();

    // ---- Combine and store transposed-fused (coalesced) ----
    float alpha = alphav[b * 8 + h];
    float* dst = fusedT + (size_t)blk * NN;
    for (int i = t; i < NN; i += 256) {
        float xg = buf[i].x * ortho;             // real part = irfft
        dst[i] = alpha * xg + (1.0f - alpha) * raw[i];
    }
}

// ---------------------------------------------------------------------------
// Kernel 4: out(B,N,DM) = x + transpose(fusedT). Coalesced via 32x33 tile.
// ---------------------------------------------------------------------------
__global__ void __launch_bounds__(256) out_transpose(const float* __restrict__ fusedT,
                                                     const float* __restrict__ x,
                                                     float* __restrict__ out) {
    __shared__ float tile[32][33];
    int b  = blockIdx.z;
    int n0 = blockIdx.x * 32;
    int c0 = blockIdx.y * 32;
    int tx = threadIdx.x, ty = threadIdx.y;
    #pragma unroll
    for (int i = 0; i < 4; i++) {
        int c = c0 + ty * 4 + i;
        int n = n0 + tx;
        tile[ty * 4 + i][tx] = fusedT[((size_t)b * DMM + c) * NN + n];
    }
    __syncthreads();
    #pragma unroll
    for (int i = 0; i < 4; i++) {
        int n = n0 + ty * 4 + i;
        int c = c0 + tx;
        size_t o = ((size_t)b * NN + n) * DMM + c;
        out[o] = x[o] + tile[tx][ty * 4 + i];
    }
}

// ---------------------------------------------------------------------------
extern "C" void kernel_launch(void* const* d_in, const int* in_sizes, int n_in,
                              void* d_out, int out_size, void* d_ws, size_t ws_size,
                              hipStream_t stream) {
    const float* x    = (const float*)d_in[0];
    const float* ln_g = (const float*)d_in[1];
    const float* ln_b = (const float*)d_in[2];
    const float* bfg  = (const float*)d_in[3];
    const float* bbg  = (const float*)d_in[4];
    const float* bfl  = (const float*)d_in[5];
    const float* bbl  = (const float*)d_in[6];
    const float* Wg1  = (const float*)d_in[7];
    const float* bg1  = (const float*)d_in[8];
    const float* Wg2  = (const float*)d_in[9];
    const float* bg2  = (const float*)d_in[10];
    const float* Wl1  = (const float*)d_in[11];
    const float* bl1  = (const float*)d_in[12];
    const float* Wl2  = (const float*)d_in[13];
    const float* bl2  = (const float*)d_in[14];
    const float* Wf1  = (const float*)d_in[15];
    const float* bf1  = (const float*)d_in[16];
    const float* Wf2  = (const float*)d_in[17];
    const float* bf2  = (const float*)d_in[18];
    float* out = (float*)d_out;

    float* ws = (float*)d_ws;
    size_t off = 0;
    float* xnT    = ws + off; off += (size_t)BB * DMM * NN;   // 64 MB
    float* fusedT = ws + off; off += (size_t)BB * DMM * NN;   // 64 MB
    float* mu     = ws + off; off += (size_t)BB * NN;
    float* rstd   = ws + off; off += (size_t)BB * NN;
    float* ctx    = ws + off; off += (size_t)16 * DMM;        // 16 rows (4 used)
    float* h1g    = ws + off; off += (size_t)16 * DMM;
    float* h1l    = ws + off; off += (size_t)16 * DMM;
    float* h1f    = ws + off; off += (size_t)16 * (DMM / 2);
    float* ag     = ws + off; off += (size_t)16 * AGC;
    float* al     = ws + off; off += (size_t)16 * 32;
    float* alphav = ws + off; off += (size_t)16 * 8;

    // Zero ctx so padded rows 4..15 feed clean zeros into the WMMA chain.
    hipMemsetAsync(ctx, 0, (size_t)16 * DMM * sizeof(float), stream);

    ln_stats<<<BB * NN, 256, 0, stream>>>(x, mu, rstd);
    dim3 tb(32, 8);
    ln_transpose<<<dim3(NN / 32, DMM / 32, BB), tb, 0, stream>>>(x, mu, rstd, ln_g, ln_b, xnT);
    ctx_sum<<<BB * DMM, 256, 0, stream>>>(xnT, ctx);

    const float invN = 1.0f / (float)NN;
    // stage-1 MLPs (gelu)
    gemm16_wmma<<<DMM / 16, 32, 0, stream>>>(ctx, invN, Wg1, bg1, h1g, DMM, DMM, 1);
    gemm16_wmma<<<DMM / 16, 32, 0, stream>>>(ctx, invN, Wl1, bl1, h1l, DMM, DMM, 1);
    gemm16_wmma<<<(DMM / 2) / 16, 32, 0, stream>>>(ctx, invN, Wf1, bf1, h1f, DMM, DMM / 2, 1);
    // stage-2 heads
    gemm16_wmma<<<(AGC + 15) / 16, 32, 0, stream>>>(h1g, 1.0f, Wg2, bg2, ag, DMM, AGC, 0);
    gemm16_wmma<<<2, 32, 0, stream>>>(h1l, 1.0f, Wl2, bl2, al, DMM, 32, 0);
    gemm16_wmma<<<1, 32, 0, stream>>>(h1f, 1.0f, Wf2, bf2, alphav, DMM / 2, 8, 2);

    // spectral + local + fuse: one 4096-pt FFT per block, 48 KB dynamic LDS
    spectral_fuse<<<BB * DMM, 256, (size_t)(NN * 3) * sizeof(float), stream>>>(
        xnT, bfg, bbg, bfl, bbl, ag, al, alphav, fusedT);

    out_transpose<<<dim3(NN / 32, DMM / 32, BB), tb, 0, stream>>>(fusedT, x, out);
}
